// STSHN_32796370272388
// MI455X (gfx1250) — matrix operand order repaired and checked by reading, not compiled, not run on hardware
//
#include <hip/hip_runtime.h>
#include <hip/hip_bf16.h>

// ---------------------------------------------------------------------------
// CDNA5 / gfx1250 implementation. wave32, WMMA f32_16x16x32_f16 everywhere the
// FLOPs live; TDM (tensor_load_to_lds) for contiguous global->LDS staging.
// Fragment layouts follow CDNA5 ISA §7.12.2; TDM descriptor per
// CDNA5 ISA.
// ---------------------------------------------------------------------------

typedef __attribute__((ext_vector_type(16))) _Float16 v16h;
typedef __attribute__((ext_vector_type(8)))  float    v8f;
typedef unsigned int u32x4 __attribute__((ext_vector_type(4)));
typedef int          i32x4 __attribute__((ext_vector_type(4)));
typedef int          i32x8 __attribute__((ext_vector_type(8)));

// Shapes (baked)
#define BB 8
#define TT 24
#define NN 256
#define CC 8
#define DD 64
#define DH 16
#define HYPER 128
#define HORIZON 12
#define CD 512                         // C*D
static const size_t NE = (size_t)BB * TT * NN * CD;   // 25,165,824 elements

__device__ __forceinline__ v8f wmma_f16(v16h a, v16h b, v8f c) {
  // (neg_a, A, neg_b, B, c_mod, C, reuse_a, reuse_b)
  return __builtin_amdgcn_wmma_f32_16x16x32_f16(false, a, false, b, (short)0, c,
                                                false, false);
}

// A fragment: logical A[16][K] row-major f16, row stride ldk (halves).
// lane L: row = L&15; elements 0..7 -> K = k0 + (L>>4)*8 + e,
//         elements 8..15 -> K = k0 + 16 + (L>>4)*8 + (e-8).   (ISA p.108)
__device__ __forceinline__ v16h frag_a_ld(const _Float16* base, int ldk, int k0) {
  const int lane = threadIdx.x & 31;
  const int row  = lane & 15;
  const int kb   = k0 + ((lane >> 4) << 3);
  const unsigned int* rp =
      reinterpret_cast<const unsigned int*>(base + (size_t)row * ldk);
  union { v16h h; unsigned int u[8]; } f;
#pragma unroll
  for (int i = 0; i < 4; ++i) f.u[i]     = rp[(kb >> 1) + i];
#pragma unroll
  for (int i = 0; i < 4; ++i) f.u[4 + i] = rp[((kb + 16) >> 1) + i];
  return f.h;
}

// B fragment: logical B[K][16]; staged COLUMN-major: bt[n][k], row stride ldk.
// lane L: col = L&15; elements 0..7 -> K = k0 + (L>>4)*8 + e,
//         elements 8..15 -> K = k0 + 16 + (L>>4)*8 + (e-8)  => 8 dword reads.
__device__ __forceinline__ v16h frag_b_ld(const _Float16* bt, int ldk, int k0) {
  const int lane = threadIdx.x & 31;
  const int col  = lane & 15;
  const int kb   = k0 + ((lane >> 4) << 3);
  const unsigned int* rp =
      reinterpret_cast<const unsigned int*>(bt + (size_t)col * ldk);
  union { v16h h; unsigned int u[8]; } f;
#pragma unroll
  for (int i = 0; i < 4; ++i) f.u[i]     = rp[(kb >> 1) + i];
#pragma unroll
  for (int i = 0; i < 4; ++i) f.u[4 + i] = rp[((kb + 16) >> 1) + i];
  return f.h;
}

__device__ __forceinline__ float lrelu(float v) {
  return (v > 0.f) ? v : 0.01f * v;
}

// ---------------------------------------------------------------------------
// Tensor Data Mover: 1-D contiguous global -> LDS DMA.
// Descriptor per CDNA5 ISA:
//  group0: [1:0]=count=1, [63:32]=lds_addr, [120:64]=global_addr, [127:126]=2
//  group1: data_size=3 (8B units), tensor_dim0/tile_dim0 = n8, dim0_stride = n8
// Issued wave-uniform; completion via TENSORcnt (s_wait_tensorcnt 0).
// ---------------------------------------------------------------------------
__device__ __forceinline__ void tdm_load_contig_to_lds(unsigned lds_byte_off,
                                                       const void* gsrc,
                                                       unsigned n8 /*8B units*/) {
  const unsigned long long ga = (unsigned long long)gsrc;
  u32x4 g0 = {};
  g0[0] = 1u;                                        // count=1 (valid, user)
  g0[1] = lds_byte_off;                              // lds_addr (bytes)
  g0[2] = (unsigned)(ga & 0xffffffffu);              // global_addr[31:0]
  g0[3] = (unsigned)((ga >> 32) & 0x01ffffffu)       // global_addr[56:32]
        | (2u << 30);                                // type=2 ("image")
  i32x8 g1 = {};
  g1[0] = (3 << 16);                                 // data_size=3 -> 8 bytes
  g1[1] = (int)((n8 & 0xffffu) << 16);               // tensor_dim0[15:0]
  g1[2] = (int)(n8 >> 16);                           // tensor_dim0[31:16]
  g1[3] = (int)((n8 & 0xffffu) << 16);               // tile_dim0
  g1[5] = (int)n8;                                   // tensor_dim0_stride[31:0]
  i32x4 z4 = {};
#if defined(__clang_major__) && (__clang_major__ >= 23)
  i32x8 z8 = {};
  __builtin_amdgcn_tensor_load_to_lds(g0, g1, z4, z4, z8, 0);
#else
  __builtin_amdgcn_tensor_load_to_lds(g0, g1, z4, z4, 0);
#endif
}

// ---------------------------------------------------------------------------
// prep: f16 operand matrices.
//  ADJT[m][n]  = adj[n][m]          (A-operand for spatial mix, 256x256)
//  HYP [m][n]  = hyper_adj[m][n]    (A-operand hyper-down, 128x256)
//  HYPT[n][m]  = hyper_adj[m][n]    (A-operand hyper-up, 256x128)
//  WTS: 6 weight mats (sQ,sK,sV,tQ,tK,tV) column-major: Wt[n][k]=W[k][n] 16x64
// ---------------------------------------------------------------------------
__global__ __launch_bounds__(256) void prep_kernel(
    const float* __restrict__ adj, const float* __restrict__ hyper,
    const float* __restrict__ sQ, const float* __restrict__ sK,
    const float* __restrict__ sV, const float* __restrict__ tQ,
    const float* __restrict__ tK, const float* __restrict__ tV,
    _Float16* __restrict__ adjT, _Float16* __restrict__ hyp,
    _Float16* __restrict__ hypT, _Float16* __restrict__ wts) {
  int idx = blockIdx.x * 256 + threadIdx.x;
  if (idx < 65536) {
    int m = idx >> 8, n = idx & 255;
    adjT[idx] = (_Float16)adj[n * 256 + m];
  } else if (idx < 98304) {
    int j = idx - 65536;
    hyp[j] = (_Float16)hyper[j];
  } else if (idx < 131072) {
    int j = idx - 98304;
    int n = j >> 7, m = j & 127;
    hypT[j] = (_Float16)hyper[m * 256 + n];
  } else if (idx < 137216) {
    int j = idx - 131072;
    int wi = j >> 10, e = j & 1023;
    int n = e >> 6, k = e & 63;
    const float* W = (wi == 0) ? sQ : (wi == 1) ? sK : (wi == 2) ? sV
                   : (wi == 3) ? tQ : (wi == 4) ? tK : tV;
    wts[j] = (_Float16)W[k * 16 + n];
  }
}

// embed[b,t,n,c,d] = x[b,t,n,c] * cat_emb[c,d]; embeds = embed
__global__ __launch_bounds__(256) void init_embed_kernel(
    const float* __restrict__ x, const float* __restrict__ cat_emb,
    float* __restrict__ embedA, float* __restrict__ embeds) {
  size_t idx = (size_t)blockIdx.x * 256 + threadIdx.x;
  size_t g = idx >> 9;
  int cd = (int)(idx & 511);
  int c = cd >> 6, d = cd & 63;
  float v = x[g * 10 + c] * cat_emb[c * 64 + d];
  embedA[idx] = v;
  embeds[idx] = v;
}

// embed = embeds * 0.5 ; embeds = embed   (between spatial and temporal stages)
__global__ __launch_bounds__(256) void half_kernel(float* __restrict__ embedA,
                                                   float* __restrict__ embeds) {
  size_t idx = (size_t)blockIdx.x * 256 + threadIdx.x;
  float v = embeds[idx] * 0.5f;
  embedA[idx] = v;
  embeds[idx] = v;
}

// ---------------------------------------------------------------------------
// Attention kernel. One wave handles TWO (b,t,n) groups packed as M=16
// (rows 0-7 = group0 categories, 8-15 = group1). Q/K/V projections are
// v_wmma_f32_16x16x32_f16 (M=16, K=64 in two steps, N=16=DH). The 8x8
// score/softmax/PV stage is scalar (one lane per token row).
//  mode 0: spatial attn -> write h (tiled x4) as f16 into out16
//  mode 1: temporal attn (q from t+1, k/v from t) -> fused lrelu + accumulate
// ---------------------------------------------------------------------------
__global__ __launch_bounds__(256) void attn_kernel(
    const float* __restrict__ srcq, const float* __restrict__ srckv,
    const _Float16* __restrict__ wts, int mode,
    _Float16* __restrict__ out16, float* __restrict__ dstA,
    float* __restrict__ embeds) {
  __shared__ _Float16 sW[3 * 1024];        // 3 weight mats col-major [16][64]
  __shared__ _Float16 sXq[8][1024];        // per-wave 16x64 query rows
  __shared__ _Float16 sXk[8][1024];        // per-wave 16x64 key/value rows
  __shared__ float    sQKV[8][3][16 * 17]; // per-wave Q,K,V 16x16 (padded)

  const int tid  = threadIdx.x;
  const int wave = tid >> 5;
  const int lane = tid & 31;

  for (int i = tid; i < 3 * 1024; i += 256) sW[i] = wts[i];

  size_t qoff[2], kvoff[2];
  const int gbase = blockIdx.x * 16 + wave * 2;
#pragma unroll
  for (int gi = 0; gi < 2; ++gi) {
    int g = gbase + gi;
    if (mode == 0) {
      qoff[gi]  = (size_t)g * CD;
      kvoff[gi] = qoff[gi];
    } else {
      int n  = g & 255;
      int bt = g >> 8;          // b*23 + tt
      int tt = bt % 23;
      int b  = bt / 23;
      qoff[gi]  = (((size_t)b * TT + tt + 1) * NN + n) * CD;
      kvoff[gi] = (((size_t)b * TT + tt)     * NN + n) * CD;
    }
  }
  // stage 16x64 f32 -> f16 for q and kv sides
  for (int idx = lane; idx < 1024; idx += 32) {
    int r = idx >> 6, d = idx & 63;
    int gi = r >> 3, c = r & 7;
    sXq[wave][idx] = (_Float16)srcq [qoff [gi] + c * 64 + d];
    sXk[wave][idx] = (_Float16)srckv[kvoff[gi] + c * 64 + d];
  }
  __syncthreads();

  const int hi = lane >> 4, col = lane & 15;
#pragma unroll
  for (int m = 0; m < 3; ++m) {           // 0:Q (from xq) 1:K 2:V (from xkv)
    const _Float16* xsrc = (m == 0) ? sXq[wave] : sXk[wave];
    v8f acc = {};
    acc = wmma_f16(frag_a_ld(xsrc, 64, 0),  frag_b_ld(sW + m * 1024, 64, 0),  acc);
    acc = wmma_f16(frag_a_ld(xsrc, 64, 32), frag_b_ld(sW + m * 1024, 64, 32), acc);
#pragma unroll
    for (int r = 0; r < 8; ++r)
      sQKV[wave][m][(r + 8 * hi) * 17 + col] = acc[r];
  }
  __syncthreads();

  if (lane < 16) {
    const int row  = lane;                 // token row (two groups of 8)
    const int goff = (row < 8) ? 0 : 8;    // block-diagonal group window
    const float* Qf = sQKV[wave][0];
    const float* Kf = sQKV[wave][1];
    const float* Vf = sQKV[wave][2];
    float s[8];
    float mx = -3.0e38f;
#pragma unroll
    for (int u = 0; u < 8; ++u) {
      float d = 0.f;
#pragma unroll
      for (int i = 0; i < 16; ++i) d += Qf[row * 17 + i] * Kf[(goff + u) * 17 + i];
      s[u] = d * 0.25f;                    // 1/sqrt(DH)
      mx = fmaxf(mx, s[u]);
    }
    float sum = 0.f;
#pragma unroll
    for (int u = 0; u < 8; ++u) { s[u] = __expf(s[u] - mx); sum += s[u]; }
    const float inv = 1.f / sum;
    float o[16];
#pragma unroll
    for (int i = 0; i < 16; ++i) o[i] = 0.f;
#pragma unroll
    for (int u = 0; u < 8; ++u) {
      const float p = s[u] * inv;
#pragma unroll
      for (int i = 0; i < 16; ++i) o[i] += p * Vf[(goff + u) * 17 + i];
    }
    const int gi = row >> 3, c = row & 7;
    const size_t ob = qoff[gi] + (size_t)c * 64;
    if (mode == 0) {
#pragma unroll
      for (int i = 0; i < 16; ++i) {
        const _Float16 hv = (_Float16)o[i];
#pragma unroll
        for (int j = 0; j < 4; ++j) out16[ob + j * 16 + i] = hv;  // tile x heads
      }
    } else {
#pragma unroll
      for (int i = 0; i < 16; ++i) {
        const float v = lrelu(o[i]);
#pragma unroll
        for (int j = 0; j < 4; ++j) {
          const size_t idx = ob + j * 16 + i;
          dstA[idx]    = v;
          embeds[idx] += v;
        }
      }
    }
  }
}

// ---------------------------------------------------------------------------
// Node-mixing matmul: per (b,t):  Out[M][512] = epilogue( A[M][K] x H[K][512] )
// Block tile 32(M) x 64(N); 8 waves = 2x4 grid of 16x16 WMMA tiles. The A
// strip (contiguous 32*K f16) is DMA'd into LDS by the Tensor Data Mover
// (overlapping the VALU transpose of the H strip); the H strip is staged
// column-major f16 with padded ldk. K is a template constant so the K-loop
// fully unrolls into back-to-back WMMAs.
//  MODE 0: Out = lrelu(acc) as f16         (spatial mix / hyper-down)
//  MODE 1: v = lrelu(acc) + h2;  dstA = v; embeds += v   (hyper-up fuse)
// ---------------------------------------------------------------------------
#define MM_LDK 264
template <int K, int MODE>
__global__ __launch_bounds__(256) void mix_mm_kernel(
    const _Float16* __restrict__ A, const _Float16* __restrict__ Hsrc,
    int Mtot, _Float16* __restrict__ out16, const _Float16* __restrict__ h2,
    float* __restrict__ dstA, float* __restrict__ embeds) {
  __shared__ _Float16 sA[32 * 256];        // A strip, row-major, stride K
  __shared__ _Float16 sB[64 * MM_LDK];     // H strip, col-major sB[c][k]

  const int tid = threadIdx.x;
  const int MB  = Mtot >> 5;
  const int bi  = blockIdx.x;
  const int bt  = bi / (MB * 8);
  const int rem = bi % (MB * 8);
  const int mb  = rem >> 3;
  const int nb  = rem & 7;
  const int n0  = nb * 64;

  // A strip via TDM: contiguous 32*K halves -> LDS offset of sA (first shared
  // object => offset 0). Wave 0 issues the DMA and drains TENSORcnt before
  // the workgroup barrier; meanwhile all waves transpose the H strip.
  if (tid < 32) {
    tdm_load_contig_to_lds(0u, A + (size_t)mb * 32 * K, (32u * K * 2u) >> 3);
    __builtin_amdgcn_s_wait_tensorcnt(0);
  }
  { // H strip: coalesced global reads, transpose into LDS
    const size_t hb = (size_t)bt * K * CD + n0;
    __builtin_prefetch(&Hsrc[hb + (size_t)(tid >> 6) * CD + (tid & 63)], 0, 3);
    for (int idx = tid; idx < K * 64; idx += 256) {
      int k = idx >> 6, c = idx & 63;
      sB[c * MM_LDK + k] = Hsrc[hb + (size_t)k * CD + c];
    }
  }
  __syncthreads();

  const int wave = tid >> 5;
  const int wm = wave >> 2, wn = wave & 3;
  v8f acc = {};
#pragma unroll
  for (int k0 = 0; k0 < K; k0 += 32) {
    acc = wmma_f16(frag_a_ld(sA + (size_t)wm * 16 * K, K, k0),
                   frag_b_ld(sB + (size_t)wn * 16 * MM_LDK, MM_LDK, k0), acc);
  }
  const int lane = tid & 31;
  const int hi = lane >> 4, col = lane & 15;
  const int mbase = mb * 32 + wm * 16 + 8 * hi;
  const int ncol  = n0 + wn * 16 + col;
  if (MODE == 0) {
    _Float16* ob = out16 + (size_t)bt * Mtot * CD;
#pragma unroll
    for (int r = 0; r < 8; ++r)
      ob[(size_t)(mbase + r) * CD + ncol] = (_Float16)lrelu(acc[r]);
  } else {
    const size_t base = (size_t)bt * NN * CD;
#pragma unroll
    for (int r = 0; r < 8; ++r) {
      const size_t idx = base + (size_t)(mbase + r) * CD + ncol;
      float v = lrelu(acc[r]) + (float)h2[idx];
      dstA[idx]    = v;
      embeds[idx] += v;
    }
  }
}

// temporal t=0 slice: dst[:, :, 0] = lrelu(src[:, :, 0]); embeds += that
__global__ __launch_bounds__(256) void t0_kernel(const float* __restrict__ src,
                                                 float* __restrict__ dst,
                                                 float* __restrict__ embeds) {
  size_t idx = (size_t)blockIdx.x * 256 + threadIdx.x;   // over B*N*512
  size_t bn = idx >> 9;
  size_t b = bn >> 8, n = bn & 255, cd = idx & 511;
  size_t off = ((b * TT) * NN + n) * CD + cd;            // t == 0
  float v = lrelu(src[off]);
  dst[off] = v;
  embeds[off] += v;
}

// final: ebar = mean_t(embeds)/2; y = ebar*w; out = y @ out_W + out_b, transposed
__global__ __launch_bounds__(128) void final_kernel(
    const float* __restrict__ embeds, const float* __restrict__ w,
    const float* __restrict__ outW, const float* __restrict__ outb,
    float* __restrict__ out) {
  __shared__ float ebar[512];
  const int b = blockIdx.x >> 8;
  const int n = blockIdx.x & 255;
  const int tid = threadIdx.x;
  const int cd0 = tid * 4;
  float a0 = 0.f, a1 = 0.f, a2 = 0.f, a3 = 0.f;
  for (int t = 0; t < TT; ++t) {
    const size_t base = (((size_t)b * TT + t) * NN + n) * CD + cd0;
    a0 += embeds[base];     a1 += embeds[base + 1];
    a2 += embeds[base + 2]; a3 += embeds[base + 3];
  }
  const float sc = 1.0f / 48.0f;   // /temporal_layers(2) then mean over T(24)
  ebar[cd0] = a0 * sc; ebar[cd0 + 1] = a1 * sc;
  ebar[cd0 + 2] = a2 * sc; ebar[cd0 + 3] = a3 * sc;
  __syncthreads();
  if (tid < CC * HORIZON) {
    const int c = tid / HORIZON, hz = tid % HORIZON;
    float s = outb[hz];
#pragma unroll 4
    for (int d = 0; d < 64; ++d)
      s += ebar[c * 64 + d] * w[c * 64 + d] * outW[d * HORIZON + hz];
    out[(((size_t)b * HORIZON + hz) * NN + n) * CC + c] = s;
  }
}

// ---------------------------------------------------------------------------
extern "C" void kernel_launch(void* const* d_in, const int* in_sizes, int n_in,
                              void* d_out, int out_size, void* d_ws,
                              size_t ws_size, hipStream_t stream) {
  (void)in_sizes; (void)n_in; (void)out_size; (void)ws_size;
  const float* x       = (const float*)d_in[0];
  const float* cat_emb = (const float*)d_in[1];
  const float* adj     = (const float*)d_in[2];
  const float* hyper   = (const float*)d_in[3];
  const float* sQ      = (const float*)d_in[4];
  const float* sK      = (const float*)d_in[5];
  const float* sV      = (const float*)d_in[6];
  const float* tQ      = (const float*)d_in[7];
  const float* tK      = (const float*)d_in[8];
  const float* tV      = (const float*)d_in[9];
  const float* w       = (const float*)d_in[10];
  const float* outW    = (const float*)d_in[11];
  const float* outb    = (const float*)d_in[12];
  float* out = (float*)d_out;

  // workspace layout (bytes): EMBED_A f32 | EMBEDS f32 | HATT f16 | H2 f16 |
  // HY f16 | small f16 mats.  EMBED_B overlays HATT+H2 (dead in temporal stage)
  char* ws = (char*)d_ws;
  float*    EMBED_A = (float*)ws;
  float*    EMBEDS  = (float*)(ws + NE * 4);
  _Float16* HATT    = (_Float16*)(ws + NE * 8);
  _Float16* H2      = (_Float16*)(ws + NE * 8 + NE * 2);
  _Float16* HY      = (_Float16*)(ws + NE * 8 + NE * 4);
  char*     sm      = ws + NE * 8 + NE * 4 + (size_t)BB * TT * HYPER * CD * 2;
  _Float16* ADJT    = (_Float16*)sm;                       // 256x256
  _Float16* HYP     = (_Float16*)(sm + 131072);            // 128x256
  _Float16* HYPT    = (_Float16*)(sm + 131072 + 65536);    // 256x128
  _Float16* WTS     = (_Float16*)(sm + 131072 + 131072);   // 6 x 16x64
  float*    EMBED_B = (float*)HATT;                        // NE f32 overlay

  const int BT = BB * TT;                                  // 192

  prep_kernel<<<536, 256, 0, stream>>>(adj, hyper, sQ, sK, sV, tQ, tK, tV,
                                       ADJT, HYP, HYPT, WTS);
  init_embed_kernel<<<(int)(NE / 256), 256, 0, stream>>>(x, cat_emb, EMBED_A,
                                                         EMBEDS);
  // ---- spatial layers ----
  for (int l = 0; l < 2; ++l) {
    attn_kernel<<<BT * NN / 16, 256, 0, stream>>>(EMBED_A, EMBED_A, WTS, 0,
                                                  HATT, nullptr, nullptr);
    // h2 = lrelu(adj^T x h)
    mix_mm_kernel<256, 0><<<BT * 8 * 8, 256, 0, stream>>>(
        ADJT, HATT, 256, H2, nullptr, nullptr, nullptr);
    // hy = lrelu(hyper_adj x h2)
    mix_mm_kernel<256, 0><<<BT * 4 * 8, 256, 0, stream>>>(
        HYP, H2, 128, HY, nullptr, nullptr, nullptr);
    // embed = lrelu(hyper_adj^T x hy) + h2 ; embeds += embed
    mix_mm_kernel<128, 1><<<BT * 8 * 8, 256, 0, stream>>>(
        HYPT, HY, 256, nullptr, H2, EMBED_A, EMBEDS);
  }
  half_kernel<<<(int)(NE / 256), 256, 0, stream>>>(EMBED_A, EMBEDS);
  // ---- temporal layers (ping-pong embed buffers) ----
  for (int l = 0; l < 2; ++l) {
    const float* src = (l == 0) ? EMBED_A : EMBED_B;
    float*       dst = (l == 0) ? EMBED_B : EMBED_A;
    t0_kernel<<<BB * NN * CD / 256, 256, 0, stream>>>(src, dst, EMBEDS);
    attn_kernel<<<BB * (TT - 1) * NN / 16, 256, 0, stream>>>(
        src, src, WTS + 3 * 1024, 1, nullptr, dst, EMBEDS);
  }
  final_kernel<<<BB * NN, 128, 0, stream>>>(EMBEDS, w, outW, outb, out);
}
